// AFNONet_32109175505280
// MI455X (gfx1250) — matrix operand order repaired and checked by reading, not compiled, not run on hardware
//
#include <hip/hip_runtime.h>
#include <hip/hip_bf16.h>

// ---------------------------------------------------------------------------
// Types for CDNA5 WMMA (wave32, 16x16x32 bf16 -> f32)
// ---------------------------------------------------------------------------
typedef __attribute__((ext_vector_type(16))) __bf16 v16bf;
typedef __attribute__((ext_vector_type(8)))  float  v8f;

__device__ __forceinline__ unsigned short f2bf(float f) {
    union { float f; unsigned u; } v; v.f = f;
    unsigned r = v.u + 0x7FFFu + ((v.u >> 16) & 1u);   // round-to-nearest-even
    return (unsigned short)(r >> 16);
}

// ---------------------------------------------------------------------------
// Generic dual-product WMMA GEMM:
//   C[b] = act( A1[b] @ B1[b] + A2[b] @ B2[b] + bias[b] + res[b] )
// A: bf16 [M][K] row-major (lda). B: bf16, either [K][N] (kn=1) or [N][K] (kn=0).
// Output f32 (outF) or bf16 (outH). act: 0 none, 1 relu, 2 relu+softshrink, 3 gelu.
// Block tile 128x64, 8 waves, each wave a 32x32 quadrant of 16x16 WMMA tiles.
// Interior tiles use CDNA5 async global->LDS b128 copies (ASYNCcnt) plus
// prefetch of the next K chunk; edge tiles use branchless clamped loads whose
// k0 is made opaque so their addressing can't be strength-reduced into
// loop-carried 64-bit adds.
// ---------------------------------------------------------------------------
#define BM 128
#define BN 64
#define LPAD 40   // halves per LDS row: 16B-aligned rows, stride 20 dwords
                  // -> conflict-free 16-lane fragment reads

union Frag { v16bf v; unsigned int u[8]; };
union U4   { uint4 q; unsigned short h[8]; };

__global__ __launch_bounds__(256) void gemm_wmma(
    int M, int N, int K,
    const unsigned short* __restrict__ A1, int lda1, long long bsA1,
    const unsigned short* __restrict__ B1, int ldb1, long long bsB1, int kn1,
    const unsigned short* __restrict__ A2, int lda2, long long bsA2,
    const unsigned short* __restrict__ B2, int ldb2, long long bsB2, int kn2,
    const float* __restrict__ bias, long long bsBias,
    const float* __restrict__ res, int ldres, long long bsRes,
    float* __restrict__ outF, unsigned short* __restrict__ outH,
    int ldc, long long bsC, int act)
{
    __shared__ __align__(16) unsigned short shA[BM * LPAD];
    __shared__ __align__(16) unsigned short shB[BN * LPAD];

    const int t    = threadIdx.x;
    const int lane = t & 31;
    const int wave = t >> 5;
    const int wm   = wave >> 1;          // 0..3 -> 32-row strip
    const int wn   = wave & 1;           // 0..1 -> 32-col strip
    const int half = lane >> 4;          // 0/1
    const int l15  = lane & 15;
    const int bm0  = blockIdx.y * BM;
    const int bn0  = blockIdx.x * BN;
    const int batch= blockIdx.z;

    const bool intM = (bm0 + BM) <= M;   // block-uniform
    const bool intN = (bn0 + BN) <= N;   // block-uniform

    v8f acc[2][2];
    for (int i = 0; i < 2; ++i)
        for (int j = 0; j < 2; ++j)
            for (int e = 0; e < 8; ++e) acc[i][j][e] = 0.0f;

    // ------- stage one product's A/B tiles into LDS (direct pointers) -------
    auto stage = [&](const unsigned short* __restrict__ Ag, int lda,
                     const unsigned short* __restrict__ Bg, int ldb, int kn,
                     int k0, bool kfull) {
        __syncthreads();
        bool usedAsync = false;
        // ---------------- A tile (BM x 32 halves) ----------------
        if (intM && kfull) {
            int row0 = t >> 2;
            int ch   = (t & 3) << 3;
            const unsigned short* s0 = Ag + (long long)(bm0 + row0) * lda + k0 + ch;
            const unsigned short* s1 = s0 + 64ll * lda;
            unsigned l0 = (unsigned)(size_t)&shA[row0 * LPAD + ch];
            unsigned l1 = (unsigned)(size_t)&shA[(row0 + 64) * LPAD + ch];
            asm volatile("global_load_async_to_lds_b128 %0, %1, off"
                         :: "v"(l0), "v"(s0) : "memory");
            asm volatile("global_load_async_to_lds_b128 %0, %1, off"
                         :: "v"(l1), "v"(s1) : "memory");
            if (k0 + 64 <= K) __builtin_prefetch(s0 + 32, 0, 3);
            usedAsync = true;
        } else {
            int k0o = k0; asm("" : "+s"(k0o));   // opaque: keep tail math local
            unsigned short va[16];
            for (int i = 0; i < 8; ++i) {          // phase 1: global loads
                int idx = t + (i << 8);
                int row = idx >> 4;
                int c2  = (idx & 15) << 1;
                int gm = bm0 + row, gk = k0o + c2;
                int gmc = gm < M ? gm : 0;
                int gkc = gk < K - 1 ? gk : 0;
                const unsigned short* src = Ag + (long long)gmc * lda + gkc;
                unsigned short h0 = src[0], h1 = src[1];
                va[2 * i]     = (gm < M && gk     < K) ? h0 : (unsigned short)0;
                va[2 * i + 1] = (gm < M && gk + 1 < K) ? h1 : (unsigned short)0;
            }
            for (int i = 0; i < 8; ++i) {          // phase 2: LDS stores
                int idx = t + (i << 8);
                int row = idx >> 4;
                int c2  = (idx & 15) << 1;
                shA[row * LPAD + c2]     = va[2 * i];
                shA[row * LPAD + c2 + 1] = va[2 * i + 1];
            }
        }
        // ------------- B tile into [n][k] LDS layout -------------
        if (kn) {            // B is [K][N]: transpose while staging
            if (intN && kfull) {
                int kk = t >> 3;             // 0..31
                int n8 = (t & 7) << 3;       // 0,8,..,56
                const unsigned short* src =
                    Bg + (long long)(k0 + kk) * ldb + bn0 + n8;
                U4 u; u.q = *(const uint4*)src;
                if (k0 + 64 <= K) __builtin_prefetch(src + 32ll * ldb, 0, 3);
                for (int j = 0; j < 8; ++j)
                    shB[(n8 + j) * LPAD + kk] = u.h[j];
            } else {
                int k0o = k0; asm("" : "+s"(k0o));
                unsigned short vb[8];
                for (int i = 0; i < 8; ++i) {      // phase 1: global loads
                    int idx = t + (i << 8);
                    int kk = idx >> 6;
                    int nn = idx & 63;
                    int gk = k0o + kk, gn = bn0 + nn;
                    int gkc = gk < K ? gk : 0;
                    int gnc = gn < N ? gn : 0;
                    unsigned short hv = Bg[(long long)gkc * ldb + gnc];
                    vb[i] = (gk < K && gn < N) ? hv : (unsigned short)0;
                }
                for (int i = 0; i < 8; ++i) {      // phase 2: LDS stores
                    int idx = t + (i << 8);
                    int kk = idx >> 6;
                    int nn = idx & 63;
                    shB[nn * LPAD + kk] = vb[i];
                }
            }
        } else {             // B is [N][K]: direct
            if (intN && kfull) {
                int row = t >> 2;
                int ch  = (t & 3) << 3;
                const unsigned short* src =
                    Bg + (long long)(bn0 + row) * ldb + k0 + ch;
                unsigned lb = (unsigned)(size_t)&shB[row * LPAD + ch];
                asm volatile("global_load_async_to_lds_b128 %0, %1, off"
                             :: "v"(lb), "v"(src) : "memory");
                if (k0 + 64 <= K) __builtin_prefetch(src + 32, 0, 3);
                usedAsync = true;
            } else {
                int k0o = k0; asm("" : "+s"(k0o));
                unsigned short vb[8];
                for (int i = 0; i < 4; ++i) {      // phase 1: global loads
                    int idx = t + (i << 8);
                    int nn = idx >> 4;
                    int c2 = (idx & 15) << 1;
                    int gn = bn0 + nn, gk = k0o + c2;
                    int gnc = gn < N ? gn : 0;
                    int gkc = gk < K - 1 ? gk : 0;
                    const unsigned short* src = Bg + (long long)gnc * ldb + gkc;
                    unsigned short h0 = src[0], h1 = src[1];
                    vb[2 * i]     = (gn < N && gk     < K) ? h0 : (unsigned short)0;
                    vb[2 * i + 1] = (gn < N && gk + 1 < K) ? h1 : (unsigned short)0;
                }
                for (int i = 0; i < 4; ++i) {      // phase 2: LDS stores
                    int idx = t + (i << 8);
                    int nn = idx >> 4;
                    int c2 = (idx & 15) << 1;
                    shB[nn * LPAD + c2]     = vb[2 * i];
                    shB[nn * LPAD + c2 + 1] = vb[2 * i + 1];
                }
            }
        }
        if (usedAsync)                       // drain this wave's async copies;
            asm volatile("s_wait_asynccnt 0" ::: "memory");
        __syncthreads();                     // then all waves' tiles resident
    };

    // ------- fragment loads (ISA 16-bit A 16x32 / B 32x16 layouts) + MMA ----
    auto mma = [&]() {
        Frag af[2], bfr[2];
        for (int i = 0; i < 2; ++i) {
            const unsigned int* arow =
                (const unsigned int*)&shA[(wm * 32 + i * 16 + l15) * LPAD];
            int b4 = half * 4;
            af[i].u[0] = arow[b4 + 0];  af[i].u[1] = arow[b4 + 1];
            af[i].u[2] = arow[b4 + 2];  af[i].u[3] = arow[b4 + 3];
            af[i].u[4] = arow[b4 + 8];  af[i].u[5] = arow[b4 + 9];
            af[i].u[6] = arow[b4 + 10]; af[i].u[7] = arow[b4 + 11];
        }
        for (int j = 0; j < 2; ++j) {
            const unsigned int* brow =
                (const unsigned int*)&shB[(wn * 32 + j * 16 + l15) * LPAD];
            int b8 = half * 4;
            for (int q = 0; q < 8; ++q) bfr[j].u[q] = brow[b8 + (q & 3) + ((q >> 2) << 3)];
        }
        for (int i = 0; i < 2; ++i)
            for (int j = 0; j < 2; ++j)
                acc[i][j] = __builtin_amdgcn_wmma_f32_16x16x32_bf16(
                    false, af[i].v, false, bfr[j].v,
                    (short)0, acc[i][j], false, false);
    };

    const bool dual = (A2 != 0);
    const unsigned short* A1g = A1 + (long long)batch * bsA1;
    const unsigned short* B1g = B1 + (long long)batch * bsB1;
    const unsigned short* A2g = dual ? A2 + (long long)batch * bsA2 : A1g;
    const unsigned short* B2g = dual ? B2 + (long long)batch * bsB2 : B1g;

    for (int k0 = 0; k0 < K; k0 += 32) {
        const bool kfull = (k0 + 32) <= K;   // block-uniform
        stage(A1g, lda1, B1g, ldb1, kn1, k0, kfull);
        mma();
        if (dual) {
            stage(A2g, lda2, B2g, ldb2, kn2, k0, kfull);
            mma();
        }
    }

    // ---- epilogue (C/D layout: lane n = l15, vgpr r -> m = r + half*8) ----
    const bool interior = intM && intN;
    for (int i = 0; i < 2; ++i)
        for (int j = 0; j < 2; ++j)
            for (int r = 0; r < 8; ++r) {
                int gm = bm0 + wm * 32 + i * 16 + r + half * 8;
                int gn = bn0 + wn * 32 + j * 16 + l15;
                if (interior || (gm < M && gn < N)) {
                    float v = acc[i][j][r];
                    if (bias) v += bias[(long long)batch * bsBias + gn];
                    if (res)  v += res[(long long)batch * bsRes + (long long)gm * ldres + gn];
                    if      (act == 1) v = fmaxf(v, 0.0f);
                    else if (act == 2) v = fmaxf(v - 0.01f, 0.0f);   // relu then softshrink
                    else if (act == 3) v = 0.5f * v * (1.0f + erff(v * 0.70710678f));
                    long long ci = (long long)batch * bsC + (long long)gm * ldc + gn;
                    if (outH) outH[ci] = f2bf(v);
                    else      outF[ci] = v;
                }
            }
}

// ---------------------------------------------------------------------------
// LayerNorm over D=256 (one token per 256-thread block); bf16 and/or f32 out,
// optional [D][P] transposed f32 store for the decoder head.
// ---------------------------------------------------------------------------
__global__ __launch_bounds__(256) void layernorm_k(
    const float* __restrict__ in, const float* __restrict__ g,
    const float* __restrict__ b, unsigned short* outH, float* outF,
    int transF, int P)
{
    int p = blockIdx.x, d = threadIdx.x;
    __shared__ float red[256];
    float v = in[(long long)p * 256 + d];
    red[d] = v; __syncthreads();
    for (int s = 128; s; s >>= 1) { if (d < s) red[d] += red[d + s]; __syncthreads(); }
    float mean = red[0] * (1.0f / 256.0f);
    __syncthreads();
    float df = v - mean;
    red[d] = df * df; __syncthreads();
    for (int s = 128; s; s >>= 1) { if (d < s) red[d] += red[d + s]; __syncthreads(); }
    float rstd = rsqrtf(red[0] * (1.0f / 256.0f) + 1e-6f);
    float o = df * rstd * g[d] + b[d];
    if (outH) outH[(long long)p * 256 + d] = f2bf(o);
    if (outF) outF[transF ? ((long long)d * P + p) : ((long long)p * 256 + d)] = o;
}

// ---------------------------------------------------------------------------
// BatchNorm + 8x8/s8 patch-embed conv + bias + pos-embed -> tokens [16200][256]
// ---------------------------------------------------------------------------
__global__ __launch_bounds__(256) void patch_embed_k(
    const float* __restrict__ x, const float* __restrict__ bw,
    const float* __restrict__ bb, const float* __restrict__ bm,
    const float* __restrict__ bv, const float* __restrict__ wT,
    const float* __restrict__ pb, const float* __restrict__ pos,
    float* __restrict__ tok)
{
    int p = blockIdx.x;               // 0..16199
    int t = threadIdx.x;              // channel d
    int hh = p / 180, wp = p % 180;
    __shared__ float patch[1280];
    for (int i = 0; i < 5; ++i) {
        int idx = t + (i << 8);
        int c  = idx >> 6;
        int ky = (idx >> 3) & 7;
        int kx = idx & 7;
        float v = x[(long long)c * 720 * 1440 + (hh * 8 + ky) * 1440 + (wp * 8 + kx)];
        v = (v - bm[c]) * rsqrtf(bv[c] + 1e-6f) * bw[c] + bb[c];
        patch[idx] = v;
    }
    __syncthreads();
    float s = pb[t] + pos[(long long)p * 256 + t];
    for (int k = 0; k < 1280; ++k) s += patch[k] * wT[k * 256 + t];
    tok[(long long)p * 256 + t] = s;
}

// ---------------------------------------------------------------------------
// ConvTranspose2d k=s=2 (+tanh, + optional inverse batch-norm)
// in [Cin][h][w], wgt [Cin][Cout][2][2], out [Cout][2h][2w]
// ---------------------------------------------------------------------------
__global__ __launch_bounds__(256) void deconv2x2_k(
    const float* __restrict__ in, const float* __restrict__ wgt,
    const float* __restrict__ bias, float* __restrict__ o,
    int Cin, int Cout, int h, int w, int doTanh,
    const float* bw, const float* bb, const float* bm, const float* bv)
{
    long long idx = (long long)blockIdx.x * 256 + threadIdx.x;
    long long tot = (long long)Cout * 4 * h * w;
    if (idx >= tot) return;
    int W2 = 2 * w;
    int oc = (int)(idx / ((long long)4 * h * w));
    long long rem = idx % ((long long)4 * h * w);
    int y = (int)(rem / W2), xq = (int)(rem % W2);
    int r = y & 1, c = xq & 1, hi = y >> 1, wi = xq >> 1;
    float s = bias[oc];
    const float* wp = wgt + (long long)oc * 4 + r * 2 + c;
    const float* ip = in + (long long)hi * w + wi;
    for (int ci = 0; ci < Cin; ++ci)
        s += ip[(long long)ci * h * w] * wp[(long long)ci * Cout * 4];
    if (doTanh) s = tanhf(s);
    if (bw) s = (s - bb[oc]) * (sqrtf(bv[oc]) + 1e-6f) / bw[oc] + bm[oc];
    o[idx] = s;
}

// ---------------------------------------------------------------------------
// Prep kernels: weight transposes / bf16 conversion / DFT matrix generation
// ---------------------------------------------------------------------------
__global__ void transpose_pw_k(const float* __restrict__ in, float* __restrict__ outp) {
    int idx = blockIdx.x * 256 + threadIdx.x;          // [256][1280] -> [1280][256]
    if (idx >= 256 * 1280) return;
    int d = idx / 1280, k = idx % 1280;
    outp[k * 256 + d] = in[idx];
}

__global__ void convT_k(const float* __restrict__ in, unsigned short* __restrict__ outp,
                        int L, int R, int C) {               // [L][R][C] -> bf16 [L][C][R]
    long long idx = (long long)blockIdx.x * 256 + threadIdx.x;
    long long tot = (long long)L * R * C;
    if (idx >= tot) return;
    int l = (int)(idx / ((long long)R * C));
    long long rem = idx % ((long long)R * C);
    int r = (int)(rem / C), c = (int)(rem % C);
    outp[(long long)l * R * C + (long long)c * R + r] = f2bf(in[idx]);
}

__global__ void conv_blk_k(const float* __restrict__ w, unsigned short* wr,
                           unsigned short* wi, unsigned short* win) {
    int idx = blockIdx.x * 256 + threadIdx.x;          // over 12*8*32*32
    if (idx >= 12 * 8192) return;
    int l = idx / 8192, r = idx % 8192;
    float a = w[(long long)l * 16384 + r];             // [l][0][...] real
    float b = w[(long long)l * 16384 + 8192 + r];      // [l][1][...] imag
    wr[idx] = f2bf(a); wi[idx] = f2bf(b); win[idx] = f2bf(-b);
}

__global__ void dft_init_k(unsigned short* Wc, unsigned short* Wsn,
                           unsigned short* Gc, unsigned short* Gs, unsigned short* Gsn,
                           unsigned short* Rc, unsigned short* Rsn) {
    int idx = blockIdx.x * 256 + threadIdx.x;
    const float TP = 6.2831853071795864f;
    const float i180 = 0.0745355992f;   // 1/sqrt(180)
    const float i90  = 0.1054092553f;   // 1/sqrt(90)
    if (idx < 96 * 192) {               // forward rfft over W: [f][w]
        int f = idx / 192, w = idx % 192;
        float c = 0.f, s = 0.f;
        if (f < 91 && w < 180) {
            float th = TP * (float)((f * w) % 180) / 180.0f;
            c = cosf(th) * i180; s = -sinf(th) * i180;
        }
        Wc[idx] = f2bf(c); Wsn[idx] = f2bf(s);
    }
    if (idx < 96 * 96) {                // symmetric 90-pt DFT cos/sin: [k][h]
        int k = idx / 96, h = idx % 96;
        float c = 0.f, s = 0.f;
        if (k < 90 && h < 90) {
            float th = TP * (float)((k * h) % 90) / 90.0f;
            c = cosf(th) * i90; s = sinf(th) * i90;
        }
        Gc[idx] = f2bf(c); Gs[idx] = f2bf(s); Gsn[idx] = f2bf(-s);
    }
    if (idx < 192 * 96) {               // inverse rfft over W: [w][f], hermitian fold
        int w = idx / 96, f = idx % 96;
        float c = 0.f, s = 0.f;
        if (w < 180 && f < 91) {
            float al = (f == 0 || f == 90) ? 1.0f : 2.0f;
            float th = TP * (float)((w * f) % 180) / 180.0f;
            c = al * cosf(th) * i180; s = -al * sinf(th) * i180;
        }
        Rc[idx] = f2bf(c); Rsn[idx] = f2bf(s);
    }
}

// ---------------------------------------------------------------------------
// Orchestration
// ---------------------------------------------------------------------------
extern "C" void kernel_launch(void* const* d_in, const int* in_sizes, int n_in,
                              void* d_out, int out_size, void* d_ws, size_t ws_size,
                              hipStream_t stream)
{
    (void)in_sizes; (void)n_in; (void)out_size; (void)ws_size;
    const float* x      = (const float*)d_in[0];
    const float* bnw    = (const float*)d_in[1];
    const float* bnb    = (const float*)d_in[2];
    const float* bnm    = (const float*)d_in[3];
    const float* bnv    = (const float*)d_in[4];
    const float* patchw = (const float*)d_in[5];
    const float* patchb = (const float*)d_in[6];
    const float* pos    = (const float*)d_in[7];
    const float* ln1w   = (const float*)d_in[8];
    const float* ln1b   = (const float*)d_in[9];
    const float* w1     = (const float*)d_in[10];
    const float* w2     = (const float*)d_in[11];
    const float* b1     = (const float*)d_in[12];
    const float* b2     = (const float*)d_in[13];
    const float* ln2w   = (const float*)d_in[14];
    const float* ln2b   = (const float*)d_in[15];
    const float* mw1    = (const float*)d_in[16];
    const float* mb1    = (const float*)d_in[17];
    const float* mw2    = (const float*)d_in[18];
    const float* mb2    = (const float*)d_in[19];
    const float* nw     = (const float*)d_in[20];
    const float* nb     = (const float*)d_in[21];
    const float* p1w    = (const float*)d_in[22];
    const float* p1b    = (const float*)d_in[23];
    const float* p2w    = (const float*)d_in[24];
    const float* p2b    = (const float*)d_in[25];
    const float* hw_    = (const float*)d_in[26];
    const float* hb_    = (const float*)d_in[27];
    float* outp = (float*)d_out;

    const int P = 16200, D = 256, F = 91, HH = 90, WW = 180;
    const int FD = F * D;                 // 23296
    const int BLK = HH * F;               // 8190 spectral positions

    char* base = (char*)d_ws;
    size_t off = 0;
    auto carve = [&](size_t bytes) -> char* {
        char* p = base + off;
        off = (off + bytes + 255) & ~(size_t)255;
        return p;
    };
    float* tokA = (float*)carve((size_t)P * D * 4);
    float* tokB = (float*)carve((size_t)P * D * 4);
    float* afno = (float*)carve((size_t)P * D * 4);
    unsigned short* lnh = (unsigned short*)carve((size_t)P * D * 2);
    size_t specB = (size_t)HH * FD * 2;                 // 90*91*256 bf16
    unsigned short* Xr = (unsigned short*)carve(specB);
    unsigned short* Xi = (unsigned short*)carve(specB);
    unsigned short* Zr = (unsigned short*)carve(specB);
    unsigned short* Zi = (unsigned short*)carve(specB);
    unsigned short* R1 = (unsigned short*)carve(specB);
    unsigned short* I1 = (unsigned short*)carve(specB);
    unsigned short* R2 = (unsigned short*)carve(specB);
    unsigned short* I2 = (unsigned short*)carve(specB);
    unsigned short* Yr = (unsigned short*)carve(specB);
    unsigned short* Yi = (unsigned short*)carve(specB);
    unsigned short* hid = (unsigned short*)carve((size_t)P * 1024 * 2);
    unsigned short* w1T = (unsigned short*)carve((size_t)12 * 1024 * 256 * 2);
    unsigned short* w2T = (unsigned short*)carve((size_t)12 * 1024 * 256 * 2);
    unsigned short* B1R = (unsigned short*)carve((size_t)12 * 8192 * 2);
    unsigned short* B1I = (unsigned short*)carve((size_t)12 * 8192 * 2);
    unsigned short* B1N = (unsigned short*)carve((size_t)12 * 8192 * 2);
    unsigned short* B2R = (unsigned short*)carve((size_t)12 * 8192 * 2);
    unsigned short* B2I = (unsigned short*)carve((size_t)12 * 8192 * 2);
    unsigned short* B2N = (unsigned short*)carve((size_t)12 * 8192 * 2);
    unsigned short* Wc  = (unsigned short*)carve(96 * 192 * 2);
    unsigned short* Wsn = (unsigned short*)carve(96 * 192 * 2);
    unsigned short* Gc  = (unsigned short*)carve(96 * 96 * 2);
    unsigned short* Gs  = (unsigned short*)carve(96 * 96 * 2);
    unsigned short* Gsn = (unsigned short*)carve(96 * 96 * 2);
    unsigned short* Rc  = (unsigned short*)carve(192 * 96 * 2);
    unsigned short* Rsn = (unsigned short*)carve(192 * 96 * 2);
    float* pwT  = (float*)carve((size_t)1280 * 256 * 4);
    float* chan = (float*)carve((size_t)256 * P * 4);
    float* dc1  = (float*)carve((size_t)320 * 180 * 360 * 4);
    float* dc2  = (float*)carve((size_t)80 * 360 * 720 * 4);

    // -------- prep (deterministic each call) --------
    transpose_pw_k<<<(256 * 1280 + 255) / 256, 256, 0, stream>>>(patchw, pwT);
    convT_k<<<(int)(((long long)12 * 256 * 1024 + 255) / 256), 256, 0, stream>>>(mw1, w1T, 12, 256, 1024);
    convT_k<<<(int)(((long long)12 * 1024 * 256 + 255) / 256), 256, 0, stream>>>(mw2, w2T, 12, 1024, 256);
    conv_blk_k<<<(12 * 8192 + 255) / 256, 256, 0, stream>>>(w1, B1R, B1I, B1N);
    conv_blk_k<<<(12 * 8192 + 255) / 256, 256, 0, stream>>>(w2, B2R, B2I, B2N);
    dft_init_k<<<(96 * 192 + 255) / 256, 256, 0, stream>>>(Wc, Wsn, Gc, Gs, Gsn, Rc, Rsn);
    patch_embed_k<<<P, 256, 0, stream>>>(x, bnw, bnb, bnm, bnv, pwT, patchb, pos, tokA);

    auto gemm = [&](int M, int N, int K, int batch,
                    const unsigned short* a1, int lda1, long long bsa1,
                    const unsigned short* bb1, int ldb1, long long bsb1, int kn1,
                    const unsigned short* a2, int lda2, long long bsa2,
                    const unsigned short* bb2, int ldb2, long long bsb2, int kn2,
                    const float* bi, long long bsbi,
                    const float* re, int ldre, long long bsre,
                    float* oF, unsigned short* oH, int ldc, long long bsc, int act) {
        dim3 g((N + BN - 1) / BN, (M + BM - 1) / BM, batch);
        gemm_wmma<<<g, 256, 0, stream>>>(M, N, K, a1, lda1, bsa1, bb1, ldb1, bsb1, kn1,
                                         a2, lda2, bsa2, bb2, ldb2, bsb2, kn2,
                                         bi, bsbi, re, ldre, bsre, oF, oH, ldc, bsc, act);
    };

    for (int L = 0; L < 12; ++L) {
        float* tin  = (L & 1) ? tokB : tokA;
        float* tout = (L & 1) ? tokA : tokB;
        // LN1 -> bf16 tokens
        layernorm_k<<<P, 256, 0, stream>>>(tin, ln1w + L * 256, ln1b + L * 256, lnh, nullptr, 0, P);
        // forward rfft over W: per-row-h GEMM, X[h][f][d]
        gemm(F, D, WW, HH, Wc, 192, 0, lnh, 256, (long long)WW * D, 1,
             nullptr, 0, 0, nullptr, 0, 0, 0, nullptr, 0, nullptr, 0, 0,
             nullptr, Xr, D, (long long)F * D, 0);
        gemm(F, D, WW, HH, Wsn, 192, 0, lnh, 256, (long long)WW * D, 1,
             nullptr, 0, 0, nullptr, 0, 0, 0, nullptr, 0, nullptr, 0, 0,
             nullptr, Xi, D, (long long)F * D, 0);
        // forward FFT over H: Zr = Gc@Xr + Gs@Xi ; Zi = Gc@Xi - Gs@Xr
        gemm(HH, FD, HH, 1, Gc, 96, 0, Xr, FD, 0, 1, Gs, 96, 0, Xi, FD, 0, 1,
             nullptr, 0, nullptr, 0, 0, nullptr, Zr, FD, 0, 0);
        gemm(HH, FD, HH, 1, Gc, 96, 0, Xi, FD, 0, 1, Gsn, 96, 0, Xr, FD, 0, 1,
             nullptr, 0, nullptr, 0, 0, nullptr, Zi, FD, 0, 0);
        // block-diagonal complex MLP, stage 1 (batched over 8 blocks, K=32)
        const unsigned short* w1r = B1R + L * 8192;
        const unsigned short* w1i = B1I + L * 8192;
        const unsigned short* w1n = B1N + L * 8192;
        const float* b1r = b1 + L * 512;  const float* b1i = b1r + 256;
        gemm(BLK, 32, 32, 8, Zr, 256, 32, w1r, 32, 1024, 1,
             Zi, 256, 32, w1n, 32, 1024, 1, b1r, 32, nullptr, 0, 0,
             nullptr, R1, 256, 32, 1);
        gemm(BLK, 32, 32, 8, Zr, 256, 32, w1i, 32, 1024, 1,
             Zi, 256, 32, w1r, 32, 1024, 1, b1i, 32, nullptr, 0, 0,
             nullptr, I1, 256, 32, 1);
        // stage 2 + relu + softshrink
        const unsigned short* w2r = B2R + L * 8192;
        const unsigned short* w2i = B2I + L * 8192;
        const unsigned short* w2n = B2N + L * 8192;
        const float* b2r = b2 + L * 512;  const float* b2i = b2r + 256;
        gemm(BLK, 32, 32, 8, R1, 256, 32, w2r, 32, 1024, 1,
             I1, 256, 32, w2n, 32, 1024, 1, b2r, 32, nullptr, 0, 0,
             nullptr, R2, 256, 32, 2);
        gemm(BLK, 32, 32, 8, R1, 256, 32, w2i, 32, 1024, 1,
             I1, 256, 32, w2r, 32, 1024, 1, b2i, 32, nullptr, 0, 0,
             nullptr, I2, 256, 32, 2);
        // inverse FFT over H: Yr = Gc@R2 - Gs@I2 ; Yi = Gs@R2 + Gc@I2
        gemm(HH, FD, HH, 1, Gc, 96, 0, R2, FD, 0, 1, Gsn, 96, 0, I2, FD, 0, 1,
             nullptr, 0, nullptr, 0, 0, nullptr, Yr, FD, 0, 0);
        gemm(HH, FD, HH, 1, Gs, 96, 0, R2, FD, 0, 1, Gc, 96, 0, I2, FD, 0, 1,
             nullptr, 0, nullptr, 0, 0, nullptr, Yi, FD, 0, 0);
        // inverse rfft over W (hermitian fold): afno = Rc@Yr + Rsn@Yi
        gemm(WW, D, F, HH, Rc, 96, 0, Yr, 256, (long long)F * D, 1,
             Rsn, 96, 0, Yi, 256, (long long)F * D, 1,
             nullptr, 0, nullptr, 0, 0, afno, nullptr, D, (long long)WW * D, 0);
        // LN2 -> bf16
        layernorm_k<<<P, 256, 0, stream>>>(afno, ln2w + L * 256, ln2b + L * 256, lnh, nullptr, 0, P);
        // MLP1: [P,256]@[256,1024] + bias, gelu -> bf16 hid
        gemm(P, 1024, 256, 1, lnh, 256, 0, w1T + (long long)L * 1024 * 256, 256, 0, 0,
             nullptr, 0, 0, nullptr, 0, 0, 0, mb1 + L * 1024, 0, nullptr, 0, 0,
             nullptr, hid, 1024, 0, 3);
        // MLP2: [P,1024]@[1024,256] + bias + residual(tin) -> f32 tout
        gemm(P, 256, 1024, 1, hid, 1024, 0, w2T + (long long)L * 256 * 1024, 1024, 0, 0,
             nullptr, 0, 0, nullptr, 0, 0, 0, mb2 + L * 256, 0, tin, 256, 0,
             tout, nullptr, 256, 0, 0);
    }

    // final LN (transposed to [D][H][W]) and decoder head
    layernorm_k<<<P, 256, 0, stream>>>(tokA, nw, nb, nullptr, chan, 1, P);
    {
        long long tot = (long long)320 * 180 * 360;
        deconv2x2_k<<<(int)((tot + 255) / 256), 256, 0, stream>>>(
            chan, p1w, p1b, dc1, 256, 320, 90, 180, 1, nullptr, nullptr, nullptr, nullptr);
    }
    {
        long long tot = (long long)80 * 360 * 720;
        deconv2x2_k<<<(int)((tot + 255) / 256), 256, 0, stream>>>(
            dc1, p2w, p2b, dc2, 320, 80, 180, 360, 1, nullptr, nullptr, nullptr, nullptr);
    }
    {
        long long tot = (long long)20 * 720 * 1440;
        deconv2x2_k<<<(int)((tot + 255) / 256), 256, 0, stream>>>(
            dc2, hw_, hb_, outp, 80, 20, 360, 720, 0, bnw, bnb, bnm, bnv);
    }
}
